// BModule_30614526886154
// MI455X (gfx1250) — matrix operand order repaired and checked by reading, not compiled, not run on hardware
//
#include <hip/hip_runtime.h>

// ---------------------------------------------------------------------------
// Problem constants (from reference): B=8, T=2048, D=512, S=2048, R=64, K=16
// ---------------------------------------------------------------------------
#define BB   8
#define TT   2048
#define DD   512
#define SS   2048
#define RR   64
#define TOPK 16
#define NT   (BB * TT)   // 16384 tokens / slot-rows

typedef __attribute__((ext_vector_type(16))) __bf16        v16bf;
typedef __attribute__((ext_vector_type(4)))  __bf16        bf16x4;
typedef __attribute__((ext_vector_type(8)))  float         v8f;
typedef __attribute__((ext_vector_type(4)))  unsigned int  u32x4;

union FragBF { v16bf v; u32x4 q[2]; };

// ---------------------------------------------------------------------------
// wave32 block reductions (blockDim.x == 256 -> 8 waves)
// ---------------------------------------------------------------------------
__device__ __forceinline__ float blockReduceSum(float v, float* sm) {
#pragma unroll
  for (int off = 16; off; off >>= 1) v += __shfl_xor(v, off, 32);
  int w = threadIdx.x >> 5, lane = threadIdx.x & 31;
  if (lane == 0) sm[w] = v;
  __syncthreads();
  float r = sm[lane & 7];   // each 8-lane group holds sm[0..7] exactly once
#pragma unroll
  for (int off = 4; off; off >>= 1) r += __shfl_xor(r, off, 32);
  __syncthreads();
  return r;
}

__device__ __forceinline__ float blockReduceMax(float v, float* sm) {
#pragma unroll
  for (int off = 16; off; off >>= 1) v = fmaxf(v, __shfl_xor(v, off, 32));
  int w = threadIdx.x >> 5, lane = threadIdx.x & 31;
  if (lane == 0) sm[w] = v;
  __syncthreads();
  float r = sm[lane & 7];
#pragma unroll
  for (int off = 4; off; off >>= 1) r = fmaxf(r, __shfl_xor(r, off, 32));
  __syncthreads();
  return r;
}

// ---------------------------------------------------------------------------
// bf16 WMMA GEMM:  C[M,N] (f32) = A[M,K] (f32) * B (f32)
//   BT == false : B given as [K,N] row-major
//   BT == true  : B given as [N,K] row-major (i.e. computes A * B^T)
// block = 256 threads = 8 waves arranged 2(M) x 4(N); block tile 32x64;
// K-step 32, double-buffered LDS (one barrier / K-step). Staging converts
// f32 -> bf16 and packs bf16x4 (ds_store_b64); globals are float4 loads.
// B is staged transposed [n][k] so both fragments are contiguous b128 reads
// matching the ISA Sec 7.12.2 wave32 lane layouts.
// ---------------------------------------------------------------------------
template <bool BT>
__global__ __launch_bounds__(256) void gemm_bf16_kernel(
    const float* __restrict__ A, const float* __restrict__ B,
    float* __restrict__ C, int Kdim, int lda, int ldb, int ldc,
    long long strideA, long long strideB, long long strideC) {
  __shared__ __align__(16) __bf16 lA[2][32][32];
  __shared__ __align__(16) __bf16 lB[2][64][32];   // [n][k]

  const float* Ab = A + (size_t)blockIdx.z * strideA;
  const float* Bb = B + (size_t)blockIdx.z * strideB;
  float*       Cb = C + (size_t)blockIdx.z * strideC;

  const int m0   = blockIdx.x * 32;
  const int n0   = blockIdx.y * 64;
  const int tid  = threadIdx.x;
  const int wave = tid >> 5, lane = tid & 31;
  const int wm   = wave >> 2, wn = wave & 3;     // 2 x 4 wave grid
  const int half = lane >> 4, l16 = lane & 15;

  auto stage = [&](int buf, int k0) {
    {  // A tile 32x32: one float4 per thread
      int r = tid >> 3, c = (tid & 7) << 2;
      float4 f = *(const float4*)&Ab[(size_t)(m0 + r) * lda + (k0 + c)];
      bf16x4 h = {(__bf16)f.x, (__bf16)f.y, (__bf16)f.z, (__bf16)f.w};
      *(bf16x4*)&lA[buf][r][c] = h;
    }
#pragma unroll
    for (int i = 0; i < 2; ++i) {  // B tile 64x32 -> LDS transposed
      int e = (tid + (i << 8)) << 2;
      int n = e >> 5, c = e & 31;
      float4 f;
      if (BT) {
        f = *(const float4*)&Bb[(size_t)(n0 + n) * ldb + (k0 + c)];
      } else {
        const float* p = &Bb[(size_t)(k0 + c) * ldb + (n0 + n)];
        f.x = p[0]; f.y = p[ldb]; f.z = p[2 * ldb]; f.w = p[3 * ldb];
      }
      bf16x4 h = {(__bf16)f.x, (__bf16)f.y, (__bf16)f.z, (__bf16)f.w};
      *(bf16x4*)&lB[buf][n][c] = h;
    }
  };

  v8f acc = {0.f, 0.f, 0.f, 0.f, 0.f, 0.f, 0.f, 0.f};

  const int nk = Kdim >> 5;
  stage(0, 0);
  __syncthreads();
  for (int kt = 0; kt < nk; ++kt) {
    const int cur = kt & 1;
    if (kt + 1 < nk) stage(cur ^ 1, (kt + 1) << 5);

    FragBF fa, fb;
    fa.q[0] = *(const u32x4*)&lA[cur][wm * 16 + l16][8 * half];
    fa.q[1] = *(const u32x4*)&lA[cur][wm * 16 + l16][16 + 8 * half];
    fb.q[0] = *(const u32x4*)&lB[cur][wn * 16 + l16][8 * half];
    fb.q[1] = *(const u32x4*)&lB[cur][wn * 16 + l16][16 + 8 * half];

    acc = __builtin_amdgcn_wmma_f32_16x16x32_bf16(
        false, fa.v, false, fb.v, (short)0, acc, false, false);
    __syncthreads();
  }

  const int cm = m0 + wm * 16 + half * 8;
  const int cn = n0 + wn * 16 + l16;
#pragma unroll
  for (int r = 0; r < 8; ++r)
    Cb[(size_t)(cm + r) * ldc + cn] = acc[r];
}

// ---------------------------------------------------------------------------
// top-16 by |value| over a 2048-row (stable: lower index wins ties, matching
// jax.lax.top_k). 256 threads, 8 contiguous values/thread (2 float4 loads),
// 16 arg-max rounds (local -> wave32 shfl -> cross-wave).
// ---------------------------------------------------------------------------
__device__ __forceinline__ void blockTopK2048(const float* __restrict__ row,
                                              float* selV, int* selI,
                                              float* sA, float* sV, int* sI) {
  const int tid = threadIdx.x;
  const float4* row4 = (const float4*)row;
  float4 f0 = row4[tid * 2];
  float4 f1 = row4[tid * 2 + 1];
  float v[8] = {f0.x, f0.y, f0.z, f0.w, f1.x, f1.y, f1.z, f1.w};
  float a[8];
#pragma unroll
  for (int i = 0; i < 8; ++i) a[i] = fabsf(v[i]);

  for (int k = 0; k < TOPK; ++k) {
    float ba = -1.f, bv = 0.f; int bi = 0x7fffffff;
#pragma unroll
    for (int i = 0; i < 8; ++i) {
      int idx = (tid << 3) + i;
      if (a[i] > ba || (a[i] == ba && idx < bi)) { ba = a[i]; bv = v[i]; bi = idx; }
    }
#pragma unroll
    for (int off = 16; off; off >>= 1) {
      float oa = __shfl_xor(ba, off, 32);
      float ov = __shfl_xor(bv, off, 32);
      int   oi = __shfl_xor(bi, off, 32);
      if (oa > ba || (oa == ba && oi < bi)) { ba = oa; bv = ov; bi = oi; }
    }
    int w = tid >> 5;
    if ((tid & 31) == 0) { sA[w] = ba; sV[w] = bv; sI[w] = bi; }
    __syncthreads();
    if (tid == 0) {
      float ga = sA[0], gv = sV[0]; int gi = sI[0];
      for (int j = 1; j < 8; ++j)
        if (sA[j] > ga || (sA[j] == ga && sI[j] < gi)) { ga = sA[j]; gv = sV[j]; gi = sI[j]; }
      selV[k] = gv; selI[k] = gi;
    }
    __syncthreads();
    int win = selI[k];
    if ((win >> 3) == tid) a[win & 7] = -1.f;   // retire the winner
    __syncthreads();
  }
}

// signed_abs_softmax over the 16 selected values -> wS[16]
__device__ __forceinline__ void softmax16Signed(const float* selV, float* wS) {
  if (threadIdx.x < TOPK) {
    float m = 0.f;
#pragma unroll
    for (int j = 0; j < TOPK; ++j) m = fmaxf(m, fabsf(selV[j]));
    float den = 0.f;
#pragma unroll
    for (int j = 0; j < TOPK; ++j) den += expf(fabsf(selV[j]) - m);
    float g  = selV[threadIdx.x];
    float sg = (g > 0.f) ? 1.f : ((g < 0.f) ? -1.f : 0.f);
    wS[threadIdx.x] = sg * expf(fabsf(g) - m) / den;
  }
  __syncthreads();
}

// ---------------------------------------------------------------------------
// small prep / norm kernels
// ---------------------------------------------------------------------------
__global__ __launch_bounds__(256) void prep_state_kernel(
    const float* __restrict__ st, float* __restrict__ statebuf) {
  const int tid = threadIdx.x;
  __shared__ float red[8];
  float v[8]; float lm = 0.f;
#pragma unroll
  for (int i = 0; i < 8; ++i) { v[i] = st[tid + (i << 8)]; lm = fmaxf(lm, fabsf(v[i])); }
  float m   = blockReduceMax(lm, red);
  float ls  = 0.f;
#pragma unroll
  for (int i = 0; i < 8; ++i) ls += expf(fabsf(v[i]) - m);
  float den = blockReduceSum(ls, red);
#pragma unroll
  for (int i = 0; i < 8; ++i) {
    float g  = v[i];
    float sg = (g > 0.f) ? 1.f : ((g < 0.f) ? -1.f : 0.f);
    float r  = sg * expf(fabsf(g) - m) / den;
    for (int b = 0; b < BB; ++b) statebuf[b * SS + tid + (i << 8)] = r;
  }
}

__global__ __launch_bounds__(256) void prep_val_kernel(
    const float* __restrict__ iv, float* __restrict__ valbuf) {
  const int s = blockIdx.x, tid = threadIdx.x;
  __shared__ float red[8];
  float2 a = ((const float2*)(iv + (size_t)s * DD))[tid];
  float ss = blockReduceSum(a.x * a.x + a.y * a.y, red);
  float inv = 1.f / (sqrtf(ss) + 1e-6f);
  float2 r = {a.x * inv, a.y * inv};
#pragma unroll
  for (int b = 0; b < BB; ++b)
    ((float2*)(valbuf + ((size_t)b * SS + s) * DD))[tid] = r;
}

__global__ __launch_bounds__(256) void row_inv_norm_kernel(
    const float* __restrict__ X, float* __restrict__ inv) {
  const int r = blockIdx.x, tid = threadIdx.x;
  __shared__ float red[8];
  float2 a = ((const float2*)(X + (size_t)r * DD))[tid];
  float ss = blockReduceSum(a.x * a.x + a.y * a.y, red);
  if (tid == 0) inv[r] = 1.f / (sqrtf(ss) + 1e-6f);
}

__global__ __launch_bounds__(256) void val_norm_kernel(float* __restrict__ valbuf) {
  const int r = blockIdx.x, tid = threadIdx.x;
  __shared__ float red[8];
  float2* d = (float2*)(valbuf + (size_t)r * DD);
  float2 a = d[tid];
  float ss  = blockReduceSum(a.x * a.x + a.y * a.y, red);
  float inv = 1.f / (sqrtf(ss) + 1e-6f);
  float2 o = {a.x * inv, a.y * inv};
  d[tid] = o;
}

__global__ __launch_bounds__(256) void state_final_kernel(
    const float* __restrict__ sb, float* __restrict__ out) {
  const int b = blockIdx.x, tid = threadIdx.x;
  __shared__ float red[8];
  const float* row = sb + b * SS;
  float v[8]; float lm = 0.f;
#pragma unroll
  for (int i = 0; i < 8; ++i) { v[i] = row[tid + (i << 8)]; lm = fmaxf(lm, fabsf(v[i])); }
  float m   = blockReduceMax(lm, red);
  float ls  = 0.f;
#pragma unroll
  for (int i = 0; i < 8; ++i) ls += expf(fabsf(v[i]) - m);
  float den = blockReduceSum(ls, red);
  float* ob = out + (size_t)b * SS * (DD + 1);
#pragma unroll
  for (int i = 0; i < 8; ++i) {
    float g  = v[i];
    float sg = (g > 0.f) ? 1.f : ((g < 0.f) ? -1.f : 0.f);
    ob[(size_t)(tid + (i << 8)) * (DD + 1)] = sg * expf(fabsf(g) - m) / den;
  }
}

// ---------------------------------------------------------------------------
// write phase: per-token top-16 -> signed softmax -> scatter-add
// ---------------------------------------------------------------------------
__global__ __launch_bounds__(256) void topk_write_kernel(
    const float* __restrict__ scores, const float* __restrict__ X,
    const float* __restrict__ xinv, float* __restrict__ statebuf,
    float* __restrict__ valbuf) {
  const int tok = blockIdx.x;
  const int b   = tok >> 11;          // T == 2048
  const int tid = threadIdx.x;
  __shared__ float sA[8], sV[8]; __shared__ int sI[8];
  __shared__ float selV[TOPK]; __shared__ int selI[TOPK];
  __shared__ float wS[TOPK];

  blockTopK2048(scores + (size_t)tok * SS, selV, selI, sA, sV, sI);
  softmax16Signed(selV, wS);

  if (tid < TOPK) {    // state_delta scatter: softplus(g)
    float g  = selV[tid];
    float sp = (g > 0.f) ? (g + log1pf(expf(-g))) : log1pf(expf(g));
    unsafeAtomicAdd(&statebuf[b * SS + selI[tid]], sp);
  }
  float2 xv = ((const float2*)(X + (size_t)tok * DD))[tid];  // d = 2*tid, 2*tid+1
  float inv = xinv[tok];
  float x0 = xv.x * inv, x1 = xv.y * inv;
#pragma unroll
  for (int k = 0; k < TOPK; ++k) {   // val_delta scatter: w * unit_norm(x)
    float* dst = valbuf + ((size_t)b * SS + selI[k]) * DD + 2 * tid;
    float  w   = wS[k];
    unsafeAtomicAdd(&dst[0], w * x0);
    unsafeAtomicAdd(&dst[1], w * x1);
  }
}

// ---------------------------------------------------------------------------
// propagation phase: per-slot top-16 edges -> gather-mix -> unit-norm -> out
// ---------------------------------------------------------------------------
__global__ __launch_bounds__(256) void edge_topk_kernel(
    const float* __restrict__ P, const float* __restrict__ val,
    float* __restrict__ out) {
  const int gs  = blockIdx.x;         // b*S + s
  const int b   = gs >> 11;           // S == 2048
  const int tid = threadIdx.x;
  __shared__ float sA[8], sV[8]; __shared__ int sI[8];
  __shared__ float selV[TOPK]; __shared__ int selI[TOPK];
  __shared__ float ew[TOPK];
  __shared__ float red[8];

  blockTopK2048(P + (size_t)gs * SS, selV, selI, sA, sV, sI);
  softmax16Signed(selV, ew);

  const float* vb = val + (size_t)b * SS * DD;
  if (tid < TOPK)   // warm neighbor rows into cache (global_prefetch_b8)
    __builtin_prefetch(vb + (size_t)selI[tid] * DD, 0, 1);

  float2 acc = ((const float2*)(val + (size_t)gs * DD))[tid];  // residual
#pragma unroll
  for (int k = 0; k < TOPK; ++k) {
    float2 t = ((const float2*)(vb + (size_t)selI[k] * DD))[tid];
    float w = ew[k];
    acc.x += w * t.x;
    acc.y += w * t.y;
  }
  float ss  = blockReduceSum(acc.x * acc.x + acc.y * acc.y, red);
  float inv = 1.f / (sqrtf(ss) + 1e-6f);
  float* orow = out + (size_t)gs * (DD + 1);  // +1 offset -> scalar stores
  orow[1 + 2 * tid] = acc.x * inv;
  orow[2 + 2 * tid] = acc.y * inv;
}

// ---------------------------------------------------------------------------
// launcher
// ---------------------------------------------------------------------------
extern "C" void kernel_launch(void* const* d_in, const int* in_sizes, int n_in,
                              void* d_out, int out_size, void* d_ws, size_t ws_size,
                              hipStream_t stream) {
  const float* x          = (const float*)d_in[0];   // [B,T,D]
  const float* init_state = (const float*)d_in[1];   // [S]
  const float* init_val   = (const float*)d_in[2];   // [S,D]
  const float* route_a    = (const float*)d_in[3];   // [D,R]
  const float* route_b    = (const float*)d_in[4];   // [R,S]
  const float* pair_a     = (const float*)d_in[5];   // [D,R]
  const float* pair_b     = (const float*)d_in[6];   // [D,R]
  float* out = (float*)d_out;                        // [B,S,1+D]

  // workspace layout (all offsets 256B aligned); p-buffer reused for scores
  char* ws = (char*)d_ws;
  float* pbuf     = (float*)(ws);                                 // 134217728 B
  float* xa       = (float*)(ws + 134217728ULL);                  // NT*R f32
  float* q        = (float*)(ws + 138412032ULL);                  // B*S*R
  float* kk       = (float*)(ws + 142606336ULL);                  // B*S*R
  float* xinv     = (float*)(ws + 146800640ULL);                  // NT
  float* statebuf = (float*)(ws + 146866176ULL);                  // B*S
  float* valbuf   = (float*)(ws + 146931712ULL);                  // B*S*D
  // total: 180,486,144 bytes

  const dim3 blk(256);

  // --- init memory level ---
  prep_state_kernel<<<1, blk, 0, stream>>>(init_state, statebuf);
  prep_val_kernel<<<SS, blk, 0, stream>>>(init_val, valbuf);
  row_inv_norm_kernel<<<NT, blk, 0, stream>>>(x, xinv);

  // --- scores = (x @ route_a) @ route_b ---
  gemm_bf16_kernel<false><<<dim3(NT / 32, RR / 64, 1), blk, 0, stream>>>(
      x, route_a, xa, DD, DD, RR, RR, 0, 0, 0);
  gemm_bf16_kernel<false><<<dim3(NT / 32, SS / 64, 1), blk, 0, stream>>>(
      xa, route_b, pbuf, RR, RR, SS, SS, 0, 0, 0);

  // --- sparse write: top-16 + signed softmax + scatter ---
  topk_write_kernel<<<NT, blk, 0, stream>>>(pbuf, x, xinv, statebuf, valbuf);
  state_final_kernel<<<BB, blk, 0, stream>>>(statebuf, out);
  val_norm_kernel<<<BB * SS, blk, 0, stream>>>(valbuf);

  // --- sparse propagation: p = (val @ pair_a) @ (val @ pair_b)^T per batch ---
  gemm_bf16_kernel<false><<<dim3(NT / 32, RR / 64, 1), blk, 0, stream>>>(
      valbuf, pair_a, q, DD, DD, RR, RR, 0, 0, 0);
  gemm_bf16_kernel<false><<<dim3(NT / 32, RR / 64, 1), blk, 0, stream>>>(
      valbuf, pair_b, kk, DD, DD, RR, RR, 0, 0, 0);
  gemm_bf16_kernel<true><<<dim3(SS / 32, SS / 64, BB), blk, 0, stream>>>(
      q, kk, pbuf, RR, RR, RR, SS,
      (long long)SS * RR, (long long)SS * RR, (long long)SS * SS);

  // --- top-16 edges + residual mix + unit-norm -> out[...,1:] ---
  edge_topk_kernel<<<BB * SS, blk, 0, stream>>>(pbuf, valbuf, out);
}
